// NonAutoRegressive_76347338653859
// MI455X (gfx1250) — matrix operand order repaired and checked by reading, not compiled, not run on hardware
//
#include <hip/hip_runtime.h>

// ---------------- problem constants ----------------
constexpr int N_NODES = 20000;
constexpr int READ_LEN = 64;
constexpr int N_EDGES = 320000;
constexpr int DIM = 128;
constexpr int KSZ = 20;
constexpr int N_LAYERS = 4;
constexpr float EPS_BN = 1e-5f;
constexpr float EPS_AGG = 1e-6f;

// ---------------- CDNA5 WMMA types ----------------
typedef __attribute__((ext_vector_type(16))) __bf16 v16bf;
typedef __attribute__((ext_vector_type(8)))  float  v8f;

// round-to-nearest-even f32 -> bf16 bits
__device__ __forceinline__ unsigned f2bf_bits(float f) {
  unsigned u = __float_as_uint(f);
  return (u + 0x7FFFu + ((u >> 16) & 1u)) >> 16;
}

// A fragment (16x32 bf16) per ISA 7.12.2: lane L(0..15)=row M, half=L>>4.
// elements e=0..7  -> K = 8*half + e ; e=8..15 -> K = 16 + 8*half + (e-8).
// Two contiguous 16B loads per lane from row-major bf16 [rows,128].
__device__ __forceinline__ v16bf load_frag_a_bf16(const unsigned short* __restrict__ base, int lane) {
  int half = lane >> 4;
  const unsigned short* p = base + (size_t)(lane & 15) * 128 + 8 * half;
  union { v16bf v; uint4 q[2]; } r;
  r.q[0] = *(const uint4*)(p);
  r.q[1] = *(const uint4*)(p + 16);
  return r.v;
}

// ---------------- WMMA GEMM: Y[r,n] = sum_k X[r,k]*W[n,k] + bias[n] ----------------
// Block = 128 threads (4 waves). The block's 128-col weight strip (128x128 bf16,
// 32 KB) is staged once in LDS (row stride padded to 272B -> conflict-minimal
// ds_load_b128), then each wave computes a 16-row x 128-col tile:
// 8 accumulator tiles, 32 v_wmma, B fragments from LDS with immediate offsets.
constexpr int LDS_ROW = 136;  // shorts per row: 128 data + 8 pad (272 B = 68 dwords, !=0 mod 64)
__global__ __launch_bounds__(128) void gemm_wmma_kernel(
    const unsigned short* __restrict__ X,   // [nrows,128] bf16
    const unsigned short* __restrict__ Wb,  // [ncols,128] bf16
    const float* __restrict__ bias,         // [ncols]
    float* __restrict__ Y, int nrows, int ldY) {
  __shared__ unsigned short wlds[128 * LDS_ROW];
  int tid = threadIdx.x;
  int lane = tid & 31;
  int wave = tid >> 5;
  int n0 = blockIdx.y * 128;

  // clamp row tile so inactive waves still make safe (redundant) loads;
  // validity enforced after the barrier to keep __syncthreads block-uniform
  int rt = blockIdx.x * 4 + wave;
  int maxrt = (nrows >> 4) - 1;
  int rtc = rt > maxrt ? maxrt : rt;
  const unsigned short* Xr = X + (size_t)rtc * 16 * 128;

  // preload the entire 16x128 A tile (8 b128 loads, overlaps LDS staging)
  v16bf a[4];
#pragma unroll
  for (int s = 0; s < 4; ++s) a[s] = load_frag_a_bf16(Xr + 32 * s, lane);

  // cooperative stage: thread tid copies weight row (n0 + tid) into LDS
  {
    const unsigned short* gsrc = Wb + (size_t)(n0 + tid) * 128;
    unsigned short* ldst = wlds + tid * LDS_ROW;
#pragma unroll
    for (int c = 0; c < 16; ++c)
      *(uint4*)(ldst + c * 8) = *(const uint4*)(gsrc + c * 8);
  }
  __syncthreads();
  if (rt > maxrt) return;  // wave-uniform exit; EXEC all-ones for WMMA below

  v8f zero = {0.f, 0.f, 0.f, 0.f, 0.f, 0.f, 0.f, 0.f};
  v8f acc[8];
#pragma unroll
  for (int t = 0; t < 8; ++t) acc[t] = zero;

  // B fragment base for this lane: col (lane&15), k offset 8*(lane>>4);
  // tile t: +16*t rows, k-step s: +32 shorts -> all immediate DS offsets
  const unsigned short* wrow = wlds + (size_t)(lane & 15) * LDS_ROW + 8 * (lane >> 4);

#pragma unroll
  for (int s = 0; s < 4; ++s) {
#pragma unroll
    for (int t = 0; t < 8; ++t) {
      const unsigned short* p = wrow + (size_t)(16 * t) * LDS_ROW + 32 * s;
      union { v16bf v; uint4 q[2]; } b;
      b.q[0] = *(const uint4*)(p);
      b.q[1] = *(const uint4*)(p + 16);
      acc[t] = __builtin_amdgcn_wmma_f32_16x16x32_bf16(false, a[s], false, b.v,
                                                       (short)0, acc[t], false, false);
    }
  }

  // D layout: lane -> col (lane&15), rows r0 + v + 8*(lane>>4)
  int r0 = rt * 16;
  int col = lane & 15;
  int mrow = r0 + 8 * (lane >> 4);
#pragma unroll
  for (int t = 0; t < 8; ++t) {
    int n = n0 + 16 * t + col;
    float bv = bias[n];
#pragma unroll
    for (int v = 0; v < 8; ++v)
      Y[(size_t)(mrow + v) * ldY + n] = acc[t][v] + bv;
  }
}

// ---------------- weight/bias conversion ----------------
__global__ void wcvt_node_kernel(const float* __restrict__ Aw, const float* __restrict__ Bw,
                                 const float* __restrict__ Dw, const float* __restrict__ Ew,
                                 unsigned short* __restrict__ Wn) {
  size_t total = (size_t)N_LAYERS * 512 * 128;
  size_t stride = (size_t)gridDim.x * blockDim.x;
  for (size_t idx = (size_t)blockIdx.x * blockDim.x + threadIdx.x; idx < total; idx += stride) {
    size_t l = idx / (512 * 128);
    size_t rem = idx % (512 * 128);
    int row = (int)(rem / 128), k = (int)(rem % 128);
    int sel = row >> 7, r = row & 127;
    const float* s = (sel == 0) ? Aw : (sel == 1) ? Bw : (sel == 2) ? Dw : Ew;
    Wn[idx] = (unsigned short)f2bf_bits(s[l * 16384 + (size_t)r * 128 + k]);
  }
}
__global__ void wcvt_edge_kernel(const float* __restrict__ Cw, unsigned short* __restrict__ Wc) {
  size_t total = (size_t)N_LAYERS * 128 * 128;
  size_t stride = (size_t)gridDim.x * blockDim.x;
  for (size_t idx = (size_t)blockIdx.x * blockDim.x + threadIdx.x; idx < total; idx += stride)
    Wc[idx] = (unsigned short)f2bf_bits(Cw[idx]);
}
__global__ void bias_stack_kernel(const float* __restrict__ Ab, const float* __restrict__ Bb,
                                  const float* __restrict__ Db, const float* __restrict__ Eb,
                                  float* __restrict__ nb) {
  int idx = blockIdx.x * blockDim.x + threadIdx.x;
  if (idx >= N_LAYERS * 512) return;
  int l = idx / 512, row = idx % 512;
  int sel = row >> 7, r = row & 127;
  const float* s = (sel == 0) ? Ab : (sel == 1) ? Bb : (sel == 2) ? Db : Eb;
  nb[idx] = s[l * 128 + r];
}

// ---------------- sequence encoder: conv1d(valid,K=20) + relu + maxpool ----------------
// writes h (f32) and its bf16 mirror h_bf
__global__ __launch_bounds__(128) void seq_encoder_kernel(
    const float* __restrict__ emb, const float* __restrict__ conv_w,
    const float* __restrict__ conv_b, const int* __restrict__ reads,
    float* __restrict__ h, unsigned short* __restrict__ h_bf) {
  __shared__ float xs[3][READ_LEN];
  int n = blockIdx.x;
  int tid = threadIdx.x;
  if (tid < READ_LEN) {
    int r = reads[(size_t)n * READ_LEN + tid];
    xs[0][tid] = emb[r * 3 + 0];
    xs[1][tid] = emb[r * 3 + 1];
    xs[2][tid] = emb[r * 3 + 2];
  }
  __syncthreads();
  int j = tid;
  float w[60];
  const float* wp = conv_w + (size_t)j * 60;
#pragma unroll
  for (int u = 0; u < 60; ++u) w[u] = wp[u];
  float bias = conv_b[j];
  float m = -3.4e38f;
  for (int p = 0; p < READ_LEN - KSZ + 1; ++p) {
    float acc = bias;
#pragma unroll
    for (int c = 0; c < 3; ++c)
#pragma unroll
      for (int t = 0; t < KSZ; ++t)
        acc = fmaf(w[c * KSZ + t], xs[c][p + t], acc);
    m = fmaxf(m, acc);
  }
  float val = fmaxf(m, 0.f);  // relu then max == max then relu
  size_t o = (size_t)n * DIM + j;
  h[o] = val;
  h_bf[o] = (unsigned short)f2bf_bits(val);
}

// ---------------- edge encoder (writes e and bf16 mirror) ----------------
__global__ void edge_enc_kernel(const float* __restrict__ sim, const float* __restrict__ len,
                                const float* __restrict__ w, const float* __restrict__ b,
                                float* __restrict__ e, unsigned short* __restrict__ e_bf) {
  size_t total = (size_t)N_EDGES * DIM;
  size_t stride = (size_t)gridDim.x * blockDim.x;
  for (size_t idx = (size_t)blockIdx.x * blockDim.x + threadIdx.x; idx < total; idx += stride) {
    size_t i = idx >> 7;
    int j = (int)(idx & 127);
    float v = sim[i] * w[2 * j] + len[i] * w[2 * j + 1] + b[j];
    e[idx] = v;
    e_bf[idx] = (unsigned short)f2bf_bits(v);
  }
}

// ---------------- zero ----------------
__global__ void zero_kernel(float* __restrict__ p, size_t n) {
  size_t stride = (size_t)gridDim.x * blockDim.x;
  for (size_t i = (size_t)blockIdx.x * blockDim.x + threadIdx.x; i < n; i += stride) p[i] = 0.f;
}

// ---------------- edge phase 1: e_hat, bn-stats, gated segment sums ----------------
// e_hat buffer already holds Ce (from GEMM). blockDim = 128 (j = feature).
constexpr int EDGES_PER_BLOCK = 64;
__global__ __launch_bounds__(128) void edge_phase1_kernel(
    float* __restrict__ e_hat, const float* __restrict__ Yn,
    const int* __restrict__ src, const int* __restrict__ dst,
    float* __restrict__ num, float* __restrict__ den, float* __restrict__ stats) {
  int j = threadIdx.x;
  size_t i0 = (size_t)blockIdx.x * EDGES_PER_BLOCK;
  float s = 0.f, q = 0.f;
  for (int k = 0; k < EDGES_PER_BLOCK; ++k) {
    size_t i = i0 + k;
    if (i >= N_EDGES) break;
    int sn = src[i], dn = dst[i];
    float v = e_hat[i * 128 + j]
            + Yn[(size_t)dn * 512 + 256 + j]   // Dh[dst]
            + Yn[(size_t)sn * 512 + 384 + j];  // Eh[src]
    e_hat[i * 128 + j] = v;
    s += v;
    q += v * v;
    float sig = 1.f / (1.f + __expf(-v));
    float bh = Yn[(size_t)sn * 512 + 128 + j];  // Bh[src]
    atomicAdd(&den[(size_t)dn * 128 + j], sig);
    atomicAdd(&num[(size_t)dn * 128 + j], sig * bh);
  }
  atomicAdd(&stats[j], s);
  atomicAdd(&stats[128 + j], q);
}

// ---------------- edge phase 2: e += relu(batchnorm(e_hat)); refresh bf16 mirror ----------------
__global__ void edge_phase2_kernel(const float* __restrict__ e_hat, float* __restrict__ e,
                                   unsigned short* __restrict__ e_bf,
                                   const float* __restrict__ stats,
                                   const float* __restrict__ g, const float* __restrict__ b,
                                   float invE) {
  size_t total = (size_t)N_EDGES * DIM;
  size_t stride = (size_t)gridDim.x * blockDim.x;
  for (size_t idx = (size_t)blockIdx.x * blockDim.x + threadIdx.x; idx < total; idx += stride) {
    int j = (int)(idx & 127);
    float mean = stats[j] * invE;
    float var = stats[128 + j] * invE - mean * mean;
    float val = (e_hat[idx] - mean) * rsqrtf(var + EPS_BN) * g[j] + b[j];
    float nv = e[idx] + fmaxf(val, 0.f);
    e[idx] = nv;
    e_bf[idx] = (unsigned short)f2bf_bits(nv);
  }
}

// ---------------- node phase 1: tmp = Ah + num/(den+eps) (in place in Yn col 0..127) + stats ----
__global__ __launch_bounds__(256) void node_phase1_kernel(
    float* __restrict__ Yn, const float* __restrict__ num, const float* __restrict__ den,
    float* __restrict__ stats) {
  size_t total = (size_t)N_NODES * DIM;
  size_t stride = (size_t)gridDim.x * blockDim.x;  // multiple of 128 (blockDim=256)
  float s = 0.f, q = 0.f;
  int j = -1;
  for (size_t idx = (size_t)blockIdx.x * blockDim.x + threadIdx.x; idx < total; idx += stride) {
    j = (int)(idx & 127);
    size_t n = idx >> 7;
    float t = Yn[n * 512 + j] + num[idx] / (den[idx] + EPS_AGG);
    Yn[n * 512 + j] = t;
    s += t;
    q += t * t;
  }
  if (j >= 0) {
    atomicAdd(&stats[256 + j], s);
    atomicAdd(&stats[384 + j], q);
  }
}

// ---------------- node phase 2: h += relu(batchnorm(tmp)); refresh bf16 mirror ----------------
__global__ void node_phase2_kernel(const float* __restrict__ Yn, float* __restrict__ h,
                                   unsigned short* __restrict__ h_bf,
                                   const float* __restrict__ stats,
                                   const float* __restrict__ g, const float* __restrict__ b,
                                   float invN) {
  size_t total = (size_t)N_NODES * DIM;
  size_t stride = (size_t)gridDim.x * blockDim.x;
  for (size_t idx = (size_t)blockIdx.x * blockDim.x + threadIdx.x; idx < total; idx += stride) {
    int j = (int)(idx & 127);
    size_t n = idx >> 7;
    float mean = stats[256 + j] * invN;
    float var = stats[384 + j] * invN - mean * mean;
    float val = (Yn[n * 512 + j] - mean) * rsqrtf(var + EPS_BN) * g[j] + b[j];
    float nv = h[idx] + fmaxf(val, 0.f);
    h[idx] = nv;
    h_bf[idx] = (unsigned short)f2bf_bits(nv);
  }
}

// ---------------- decoder: out[i] = dec_w . [h[src], h[dst], e] + dec_b ----------------
__global__ __launch_bounds__(128) void decoder_kernel(
    const float* __restrict__ h, const float* __restrict__ e,
    const int* __restrict__ src, const int* __restrict__ dst,
    const float* __restrict__ dw, const float* __restrict__ db, float* __restrict__ out) {
  int lane = threadIdx.x & 31;
  int wave = threadIdx.x >> 5;
  size_t i = (size_t)blockIdx.x * 4 + wave;
  if (i >= N_EDGES) return;
  const float* hs = h + (size_t)src[i] * 128;
  const float* hd = h + (size_t)dst[i] * 128;
  const float* er = e + i * 128;
  float acc = 0.f;
  for (int j = lane; j < 128; j += 32)
    acc += dw[j] * hs[j] + dw[128 + j] * hd[j] + dw[256 + j] * er[j];
#pragma unroll
  for (int o = 16; o > 0; o >>= 1) acc += __shfl_down(acc, o, 32);
  if (lane == 0) out[i] = acc + db[0];
}

// ---------------- host launcher ----------------
static inline int gs_blocks(size_t n) {
  size_t b = (n + 255) / 256;
  return (int)(b < 16384 ? b : 16384);
}

extern "C" void kernel_launch(void* const* d_in, const int* in_sizes, int n_in,
                              void* d_out, int out_size, void* d_ws, size_t ws_size,
                              hipStream_t stream) {
  const float* emb    = (const float*)d_in[0];
  const float* conv_w = (const float*)d_in[1];
  const float* conv_b = (const float*)d_in[2];
  const float* ee_w   = (const float*)d_in[3];
  const float* ee_b   = (const float*)d_in[4];
  const float* Aw = (const float*)d_in[5];
  const float* Ab = (const float*)d_in[6];
  const float* Bw = (const float*)d_in[7];
  const float* Bb = (const float*)d_in[8];
  const float* Cw = (const float*)d_in[9];
  const float* Cb = (const float*)d_in[10];
  const float* Dw = (const float*)d_in[11];
  const float* Db = (const float*)d_in[12];
  const float* Ew = (const float*)d_in[13];
  const float* Eb = (const float*)d_in[14];
  const float* bnh_g = (const float*)d_in[15];
  const float* bnh_b = (const float*)d_in[16];
  const float* bne_g = (const float*)d_in[17];
  const float* bne_b = (const float*)d_in[18];
  const float* dec_w = (const float*)d_in[19];
  const float* dec_b = (const float*)d_in[20];
  const float* sim    = (const float*)d_in[21];
  const float* length = (const float*)d_in[22];
  const int* reads = (const int*)d_in[23];
  const int* src   = (const int*)d_in[24];
  const int* dst   = (const int*)d_in[25];
  float* out = (float*)d_out;

  // workspace carve-out (256B aligned)
  char* ws = (char*)d_ws;
  size_t off = 0;
  auto alloc = [&](size_t bytes) -> void* {
    off = (off + 255) & ~(size_t)255;
    void* p = ws + off;
    off += bytes;
    return p;
  };
  float* h     = (float*)alloc((size_t)N_NODES * 128 * 4);
  float* e     = (float*)alloc((size_t)N_EDGES * 128 * 4);
  float* Yn    = (float*)alloc((size_t)N_NODES * 512 * 4);   // Ah|Bh|Dh|Eh, then tmp in col 0..127
  float* e_hat = (float*)alloc((size_t)N_EDGES * 128 * 4);   // Ce, then e_hat
  float* num   = (float*)alloc((size_t)N_NODES * 128 * 4);
  float* den   = (float*)alloc((size_t)N_NODES * 128 * 4);
  unsigned short* h_bf = (unsigned short*)alloc((size_t)N_NODES * 128 * 2);
  unsigned short* e_bf = (unsigned short*)alloc((size_t)N_EDGES * 128 * 2);
  unsigned short* Wn = (unsigned short*)alloc((size_t)N_LAYERS * 512 * 128 * 2);
  unsigned short* Wc = (unsigned short*)alloc((size_t)N_LAYERS * 128 * 128 * 2);
  float* nbias = (float*)alloc((size_t)N_LAYERS * 512 * 4);
  float* stats = (float*)alloc(512 * 4);  // [esum|esq|nsum|nsq] x128

  // weights -> bf16 (stacked A|B|D|E), biases stacked
  wcvt_node_kernel<<<512, 256, 0, stream>>>(Aw, Bw, Dw, Ew, Wn);
  wcvt_edge_kernel<<<256, 256, 0, stream>>>(Cw, Wc);
  bias_stack_kernel<<<(N_LAYERS * 512 + 255) / 256, 256, 0, stream>>>(Ab, Bb, Db, Eb, nbias);

  // encoders
  seq_encoder_kernel<<<N_NODES, 128, 0, stream>>>(emb, conv_w, conv_b, reads, h, h_bf);
  edge_enc_kernel<<<gs_blocks((size_t)N_EDGES * 128), 256, 0, stream>>>(sim, length, ee_w, ee_b, e, e_bf);

  const float invE = 1.f / (float)N_EDGES;
  const float invN = 1.f / (float)N_NODES;

  for (int l = 0; l < N_LAYERS; ++l) {
    // node GEMM: Yn[N,512] = h @ [Aw|Bw|Dw|Ew]^T + bias   (1250 row tiles, 4 col strips of 128)
    {
      dim3 grid((N_NODES / 16 + 3) / 4, 512 / 128);
      gemm_wmma_kernel<<<grid, 128, 0, stream>>>(h_bf, Wn + (size_t)l * 512 * 128,
                                                 nbias + (size_t)l * 512, Yn, N_NODES, 512);
    }
    // edge GEMM: e_hat = Ce = e @ Cw^T + Cb   (20000 row tiles, 1 col strip)
    {
      dim3 grid((N_EDGES / 16) / 4, 1);
      gemm_wmma_kernel<<<grid, 128, 0, stream>>>(e_bf, Wc + (size_t)l * 128 * 128,
                                                 Cb + (size_t)l * 128, e_hat, N_EDGES, 128);
    }
    // zero per-layer accumulators
    zero_kernel<<<gs_blocks((size_t)N_NODES * 128), 256, 0, stream>>>(num, (size_t)N_NODES * 128);
    zero_kernel<<<gs_blocks((size_t)N_NODES * 128), 256, 0, stream>>>(den, (size_t)N_NODES * 128);
    zero_kernel<<<1, 256, 0, stream>>>(stats, 512);

    edge_phase1_kernel<<<(N_EDGES + EDGES_PER_BLOCK - 1) / EDGES_PER_BLOCK, 128, 0, stream>>>(
        e_hat, Yn, src, dst, num, den, stats);
    edge_phase2_kernel<<<gs_blocks((size_t)N_EDGES * 128), 256, 0, stream>>>(
        e_hat, e, e_bf, stats, bne_g + l * 128, bne_b + l * 128, invE);
    node_phase1_kernel<<<gs_blocks((size_t)N_NODES * 128), 256, 0, stream>>>(Yn, num, den, stats);
    node_phase2_kernel<<<gs_blocks((size_t)N_NODES * 128), 256, 0, stream>>>(
        Yn, h, h_bf, stats, bnh_g + l * 128, bnh_b + l * 128, invN);
  }

  decoder_kernel<<<(N_EDGES + 3) / 4, 128, 0, stream>>>(h, e, src, dst, dec_w, dec_b, out);
}
